// MaxPooled_41549513622109
// MI455X (gfx1250) — compile-verified, matrix operand order checked by reading, not run on hardware
//
#include <hip/hip_runtime.h>
#include <stdint.h>

// ---------------- CDNA5 types ----------------
typedef __attribute__((ext_vector_type(16))) __bf16         v16bf;
typedef __attribute__((ext_vector_type(8)))  float          v8f;
typedef __attribute__((ext_vector_type(8)))  unsigned short v8u;

union BF16x16 { v8u u[2]; v16bf v; };

__device__ __forceinline__ uint16_t f2bf(float f) {
    uint32_t u = __float_as_uint(f);
    uint32_t r = (u + 0x7FFFu + ((u >> 16) & 1u)) >> 16;   // RNE
    return (uint16_t)r;
}
__device__ __forceinline__ float bf2f(uint16_t u) {
    return __uint_as_float(((uint32_t)u) << 16);
}

// Problem constants
#define SPP   8
#define HH    96
#define NF    64
#define NG    3
#define WIDTH 128
#define KS    21
#define KK    441          // 21*21
#define PH    76           // final spatial
#define CP0   96           // padded input channels (67 -> 96)

// ---------------------------------------------------------------------------
// Pack emb_in: [s][y][x][c] bf16, c: 0..2 = global feats, 3..66 = features, rest 0
// ---------------------------------------------------------------------------
__global__ void prep_input_kernel(const float* __restrict__ feat,
                                  const float* __restrict__ gf,
                                  uint16_t* __restrict__ dst) {
    size_t id = (size_t)blockIdx.x * blockDim.x + threadIdx.x;
    size_t total = (size_t)SPP * HH * HH * CP0;
    if (id >= total) return;
    int c   = (int)(id % CP0);
    size_t pix = id / CP0;
    int x = (int)(pix % HH);
    int y = (int)((pix / HH) % HH);
    int n = (int)(pix / (HH * HH));
    float v = 0.f;
    if (c < NG)            v = gf[c];
    else if (c < NG + NF)  v = feat[(((size_t)n * NF + (c - NG)) * HH + y) * HH + x];
    dst[id] = f2bf(v);
}

// ---------------------------------------------------------------------------
// Pack weights (OIHW f32) into per-lane WMMA-B register image, bf16.
// wp index = (((r*nck + ck)*ntiles + nt)*32 + lane)*16 + e
//   lane: col N = lane&15 ; base = (lane&16)?8:0
//   e:    kk = base + ((e&8)?16:0) + (e&7) ; ci = ck*32 + kk ; co = nt*16 + N
// ---------------------------------------------------------------------------
__global__ void pack_weights_kernel(const float* __restrict__ w,
                                    uint16_t* __restrict__ dst,
                                    int Cout, int Cin, int ntiles, int nck) {
    size_t id = (size_t)blockIdx.x * blockDim.x + threadIdx.x;
    size_t total = (size_t)9 * nck * ntiles * 512;
    if (id >= total) return;
    int e    = (int)(id & 15);
    size_t t = id >> 4;
    int lane = (int)(t & 31); t >>= 5;
    int nt   = (int)(t % ntiles); t /= ntiles;
    int ck   = (int)(t % nck);
    int r    = (int)(t / nck);
    int base = (lane & 16) ? 8 : 0;
    int kk   = base + ((e & 8) ? 16 : 0) + (e & 7);
    int ci   = ck * 32 + kk;
    int co   = nt * 16 + (lane & 15);
    float v = 0.f;
    if (co < Cout && ci < Cin)
        v = w[((size_t)co * Cin + ci) * 9 + r];     // OIHW, r = ky*3+kx
    dst[id] = f2bf(v);
}

// ---------------------------------------------------------------------------
// Implicit-GEMM 3x3 VALID conv via v_wmma_f32_16x16x32_bf16.
// NCK = Cpad/32 is a template constant -> the 9*NCK K-loop fully unrolls.
// Each wave: M = 32 output pixels (two 16x16 accumulators sharing one B tile),
//            N = 16 output channels. Block = 8 waves -> 32 px x 128 ch.
// act: 0=none 1=leaky(0.01) 2=relu ; outF32: store f32 else bf16.
// ---------------------------------------------------------------------------
template <int NCK>
__global__ __launch_bounds__(256)
void conv_wmma_kernel(const uint16_t* __restrict__ in,
                      const uint16_t* __restrict__ wp,
                      const float* __restrict__ bias,
                      void* __restrict__ outv,
                      int Hin, int Win,
                      int Hout, int Wout, int Nimg,
                      int Cout, int ntiles,
                      int act, int outF32) {
    constexpr int CPAD = NCK * 32;
    const int lane = threadIdx.x & 31;
    const int wave = threadIdx.x >> 5;
    const int nt   = blockIdx.y * 8 + wave;
    if (nt >= ntiles) return;                      // wave-uniform skip

    const int totPix = Nimg * Hout * Wout;
    const int ptile  = blockIdx.x * 32;
    int p0 = ptile + (lane & 15);
    int p1 = p0 + 16;
    if (p0 >= totPix) p0 = totPix - 1;             // clamp, masked at store
    if (p1 >= totPix) p1 = totPix - 1;
    const int x0 = p0 % Wout, y0 = (p0 / Wout) % Hout, n0 = p0 / (Wout * Hout);
    const int x1 = p1 % Wout, y1 = (p1 / Wout) % Hout, n1 = p1 / (Wout * Hout);
    const int base = (lane & 16) ? 8 : 0;          // A-operand K-half select

    const int co = nt * 16 + (lane & 15);
    const float bv = bias[co < Cout ? co : (Cout - 1)];
    v8f acc0, acc1;
#pragma unroll
    for (int j = 0; j < 8; ++j) { acc0[j] = bv; acc1[j] = bv; }

#pragma unroll
    for (int r = 0; r < 9; ++r) {
        const int ky = r / 3, kx = r % 3;
        const uint16_t* ap0 =
            in + (((size_t)(n0 * Hin + y0 + ky)) * Win + (x0 + kx)) * CPAD + base;
        const uint16_t* ap1 =
            in + (((size_t)(n1 * Hin + y1 + ky)) * Win + (x1 + kx)) * CPAD + base;
        const uint16_t* bp =
            wp + (((size_t)r * NCK) * ntiles + nt) * 512 + (size_t)lane * 16;
        if (r < 8)  // prefetch next tap's packed weights (global_prefetch_b8)
            __builtin_prefetch(wp + (((size_t)(r + 1) * NCK) * ntiles + nt) * 512 +
                                   (size_t)lane * 16, 0, 1);
#pragma unroll
        for (int ck = 0; ck < NCK; ++ck) {
            BF16x16 a0, a1, b;
            a0.u[0] = *(const v8u*)(ap0);          // channels base..base+7
            a0.u[1] = *(const v8u*)(ap0 + 16);     // channels base+16..base+23
            a1.u[0] = *(const v8u*)(ap1);
            a1.u[1] = *(const v8u*)(ap1 + 16);
            b.u[0]  = *(const v8u*)(bp);
            b.u[1]  = *(const v8u*)(bp + 8);
            acc0 = __builtin_amdgcn_wmma_f32_16x16x32_bf16(
                       false, a0.v, false, b.v, (short)0, acc0, false, false);
            acc1 = __builtin_amdgcn_wmma_f32_16x16x32_bf16(
                       false, a1.v, false, b.v, (short)0, acc1, false, false);
            ap0 += 32; ap1 += 32;
            bp  += (size_t)ntiles * 512;
        }
    }

    // D layout: lane<16 -> N=lane, M=j ; lane>=16 -> N=lane-16, M=j+8
#pragma unroll
    for (int t = 0; t < 2; ++t) {
        v8f acc = t ? acc1 : acc0;
        const int pb = ptile + t * 16 + ((lane >> 4) << 3);
#pragma unroll
        for (int j = 0; j < 8; ++j) {
            float v = acc[j];
            if (act == 1)      v = (v > 0.f) ? v : 0.01f * v;
            else if (act == 2) v = fmaxf(v, 0.f);
            const int pj = pb + j;
            if (pj < totPix && co < Cout) {
                const size_t o = (size_t)pj * Cout + co;
                if (outF32) ((float*)outv)[o] = v;
                else        ((uint16_t*)outv)[o] = f2bf(v);
            }
        }
    }
}

// ---------------------------------------------------------------------------
// reg input: [s][y][x][c] bf16, 82x82, c 0..63 = cropped features,
// 64..191 = max-over-spp of emb output (bufA: [s][82][82][128] bf16)
// ---------------------------------------------------------------------------
__global__ void build_reg_in_kernel(const float* __restrict__ feat,
                                    const uint16_t* __restrict__ emb,
                                    uint16_t* __restrict__ dst) {
    const int P = 82, C = 192;
    size_t id = (size_t)blockIdx.x * blockDim.x + threadIdx.x;
    size_t total = (size_t)SPP * P * P * C;
    if (id >= total) return;
    int c = (int)(id % C);
    size_t pix = id / C;
    int x = (int)(pix % P);
    int y = (int)((pix / P) % P);
    int n = (int)(pix / (P * P));
    uint16_t o;
    if (c < NF) {
        o = f2bf(feat[(((size_t)n * NF + c) * HH + (y + 7)) * HH + (x + 7)]);
    } else {
        const int cc = c - NF;
        float m = -3.402823466e38f;
        for (int s = 0; s < SPP; ++s)
            m = fmaxf(m, bf2f(emb[(((size_t)s * P + y) * P + x) * WIDTH + cc]));
        o = f2bf(m);
    }
    dst[id] = o;
}

// ---------------------------------------------------------------------------
// Global max of logits, two-pass LDS tree reduction
// ---------------------------------------------------------------------------
__global__ void reduce_max1_kernel(const float* __restrict__ x, size_t n,
                                   float* __restrict__ partial) {
    __shared__ float sm[256];
    float m = -3.402823466e38f;
    for (size_t i = (size_t)blockIdx.x * blockDim.x + threadIdx.x; i < n;
         i += (size_t)gridDim.x * blockDim.x)
        m = fmaxf(m, x[i]);
    sm[threadIdx.x] = m;
    __syncthreads();
    for (int s = 128; s > 0; s >>= 1) {
        if (threadIdx.x < s) sm[threadIdx.x] = fmaxf(sm[threadIdx.x], sm[threadIdx.x + s]);
        __syncthreads();
    }
    if (threadIdx.x == 0) partial[blockIdx.x] = sm[0];
}

__global__ void reduce_max2_kernel(const float* __restrict__ partial, int n,
                                   float* __restrict__ out) {
    __shared__ float sm[256];
    float m = -3.402823466e38f;
    for (int i = threadIdx.x; i < n; i += 256) m = fmaxf(m, partial[i]);
    sm[threadIdx.x] = m;
    __syncthreads();
    for (int s = 128; s > 0; s >>= 1) {
        if (threadIdx.x < s) sm[threadIdx.x] = fmaxf(sm[threadIdx.x], sm[threadIdx.x + s]);
        __syncthreads();
    }
    if (threadIdx.x == 0) out[0] = sm[0];
}

// ---------------------------------------------------------------------------
// Fused softmax-weighted 21x21 kernel apply on radiance.
// logits: [ (s*76+y)*76+x ][441] f32 ; rad: (8,3,96,96) f32
// ---------------------------------------------------------------------------
__global__ void apply_kernel(const float* __restrict__ logits,
                             const float* __restrict__ rad,
                             const float* __restrict__ mptr,
                             float* __restrict__ out) {
    int pix = blockIdx.x * blockDim.x + threadIdx.x;
    if (pix >= PH * PH) return;
    const int y = pix / PH, x = pix % PH;
    const float m = mptr[0];
    float a0 = 0.f, a1 = 0.f, a2 = 0.f, sw = 0.f;
    for (int s = 0; s < SPP; ++s) {
        const float* lg = logits + ((size_t)(s * PH + y) * PH + x) * KK;
        const float* r0 = rad + (((size_t)(s * 3 + 0) * HH + y) * HH + x);
        const float* r1 = rad + (((size_t)(s * 3 + 1) * HH + y) * HH + x);
        const float* r2 = rad + (((size_t)(s * 3 + 2) * HH + y) * HH + x);
        for (int ky = 0; ky < KS; ++ky) {
            const int ro = ky * HH;
            for (int kx = 0; kx < KS; ++kx) {
                const float w = __expf(lg[ky * KS + kx] - m);
                sw += w;
                a0 = fmaf(w, r0[ro + kx], a0);
                a1 = fmaf(w, r1[ro + kx], a1);
                a2 = fmaf(w, r2[ro + kx], a2);
            }
        }
    }
    const float inv = 1.f / (sw + 1e-8f);
    out[0 * PH * PH + pix] = a0 * inv;
    out[1 * PH * PH + pix] = a1 * inv;
    out[2 * PH * PH + pix] = a2 * inv;
}

// ---------------------------------------------------------------------------
// Host side
// ---------------------------------------------------------------------------
static inline size_t alignup(size_t v) { return (v + 255) & ~(size_t)255; }

static void launch_conv(int nck, dim3 grid, hipStream_t stream,
                        const uint16_t* in, const uint16_t* wp, const float* b,
                        void* out, int Hin, int Hout, int Nimg,
                        int Cout, int ntiles, int act, int outF32) {
    switch (nck) {
    case 3: conv_wmma_kernel<3><<<grid, 256, 0, stream>>>(in, wp, b, out, Hin, Hin,
                Hout, Hout, Nimg, Cout, ntiles, act, outF32); break;
    case 4: conv_wmma_kernel<4><<<grid, 256, 0, stream>>>(in, wp, b, out, Hin, Hin,
                Hout, Hout, Nimg, Cout, ntiles, act, outF32); break;
    case 6: conv_wmma_kernel<6><<<grid, 256, 0, stream>>>(in, wp, b, out, Hin, Hin,
                Hout, Hout, Nimg, Cout, ntiles, act, outF32); break;
    }
}

extern "C" void kernel_launch(void* const* d_in, const int* in_sizes, int n_in,
                              void* d_out, int out_size, void* d_ws, size_t ws_size,
                              hipStream_t stream) {
    (void)in_sizes; (void)n_in; (void)out_size; (void)ws_size;
    const float* feat = (const float*)d_in[0];
    const float* rad  = (const float*)d_in[1];
    const float* gf   = (const float*)d_in[2];

    // ---- workspace carve-up ----
    char* ws = (char*)d_ws;
    size_t off = 0;
    auto take = [&](size_t bytes) { size_t r = off; off = alignup(off + bytes); return r; };
    const size_t o_embin  = take((size_t)SPP * HH * HH * CP0 * 2);
    const size_t o_bufA   = take((size_t)SPP * 94 * 94 * WIDTH * 2);
    const size_t o_bufB   = take((size_t)SPP * 94 * 94 * WIDTH * 2);
    const size_t o_regin  = take((size_t)SPP * 82 * 82 * 192 * 2);
    const size_t o_logits = take((size_t)SPP * PH * PH * KK * 4);
    const size_t o_part   = take((size_t)4096 * 4);
    const size_t o_mval   = take(4);

    uint16_t* embin  = (uint16_t*)(ws + o_embin);
    uint16_t* bufA   = (uint16_t*)(ws + o_bufA);
    uint16_t* bufB   = (uint16_t*)(ws + o_bufB);
    uint16_t* regin  = (uint16_t*)(ws + o_regin);
    float*    logits = (float*)   (ws + o_logits);
    float*    part   = (float*)   (ws + o_part);
    float*    mval   = (float*)   (ws + o_mval);

    // ---- layer configuration ----
    struct Cfg { const float* w; const float* b; int Cout, Cin, Cpad, ntiles, nck; uint16_t* wp; };
    Cfg cfg[10];
    for (int i = 0; i < 7; ++i) {
        cfg[i].w = (const float*)d_in[3 + 2 * i];
        cfg[i].b = (const float*)d_in[4 + 2 * i];
        cfg[i].Cout = WIDTH;
        cfg[i].Cin  = (i == 0) ? (NF + NG) : WIDTH;
        cfg[i].Cpad = (i == 0) ? CP0 : WIDTH;
        cfg[i].ntiles = WIDTH / 16;
    }
    for (int j = 0; j < 3; ++j) {
        cfg[7 + j].w = (const float*)d_in[17 + 2 * j];
        cfg[7 + j].b = (const float*)d_in[18 + 2 * j];
    }
    cfg[7].Cout = WIDTH; cfg[7].Cin = 192;   cfg[7].Cpad = 192;   cfg[7].ntiles = 8;
    cfg[8].Cout = WIDTH; cfg[8].Cin = WIDTH; cfg[8].Cpad = WIDTH; cfg[8].ntiles = 8;
    cfg[9].Cout = KK;    cfg[9].Cin = WIDTH; cfg[9].Cpad = WIDTH; cfg[9].ntiles = 28;
    for (int i = 0; i < 10; ++i) {
        cfg[i].nck = cfg[i].Cpad / 32;
        cfg[i].wp  = (uint16_t*)(ws + take((size_t)9 * cfg[i].nck * cfg[i].ntiles * 512 * 2));
    }

    // ---- 1) pack input + weights ----
    {
        size_t tot = (size_t)SPP * HH * HH * CP0;
        prep_input_kernel<<<(unsigned)((tot + 255) / 256), 256, 0, stream>>>(feat, gf, embin);
    }
    for (int i = 0; i < 10; ++i) {
        size_t tot = (size_t)9 * cfg[i].nck * cfg[i].ntiles * 512;
        pack_weights_kernel<<<(unsigned)((tot + 255) / 256), 256, 0, stream>>>(
            cfg[i].w, cfg[i].wp, cfg[i].Cout, cfg[i].Cin, cfg[i].ntiles, cfg[i].nck);
    }

    // ---- 2) emb chain (7 WMMA convs, ping-pong) ----
    const uint16_t* cin_ptr = embin;
    uint16_t* bufs[2] = { bufA, bufB };
    for (int i = 0; i < 7; ++i) {
        const int Hin = HH - 2 * i, Hout = Hin - 2;
        uint16_t* outp = bufs[i & 1];
        const int totPix = SPP * Hout * Hout;
        dim3 grid((totPix + 31) / 32, (cfg[i].ntiles + 7) / 8);
        launch_conv(cfg[i].nck, grid, stream, cin_ptr, cfg[i].wp, cfg[i].b, outp,
                    Hin, Hout, SPP, cfg[i].Cout, cfg[i].ntiles,
                    (i < 6) ? 1 : 2, /*outF32=*/0);
        cin_ptr = outp;
    }
    // emb output lives in bufA (layer index 6 -> bufs[0])

    // ---- 3) spp-max aggregation + concat with cropped features ----
    {
        size_t tot = (size_t)SPP * 82 * 82 * 192;
        build_reg_in_kernel<<<(unsigned)((tot + 255) / 256), 256, 0, stream>>>(feat, bufA, regin);
    }

    // ---- 4) reg chain ----
    {
        int totPix = SPP * 80 * 80;
        dim3 g((totPix + 31) / 32, 1);
        launch_conv(6, g, stream, regin, cfg[7].wp, cfg[7].b, bufB,
                    82, 80, SPP, WIDTH, 8, 1, 0);
    }
    {
        int totPix = SPP * 78 * 78;
        dim3 g((totPix + 31) / 32, 1);
        launch_conv(4, g, stream, bufB, cfg[8].wp, cfg[8].b, bufA,
                    80, 78, SPP, WIDTH, 8, 1, 0);
    }
    {
        int totPix = SPP * PH * PH;
        dim3 g((totPix + 31) / 32, (28 + 7) / 8);
        launch_conv(4, g, stream, bufA, cfg[9].wp, cfg[9].b, logits,
                    78, PH, SPP, KK, 28, 0, /*outF32=*/1);
    }

    // ---- 5) global max + fused softmax apply ----
    const size_t nlog = (size_t)SPP * PH * PH * KK;
    reduce_max1_kernel<<<4096, 256, 0, stream>>>(logits, nlog, part);
    reduce_max2_kernel<<<1, 256, 0, stream>>>(part, 4096, mval);
    apply_kernel<<<(PH * PH + 255) / 256, 256, 0, stream>>>(logits, rad, mval, (float*)d_out);
}
